// MultiHeadAttention_25847113187553
// MI455X (gfx1250) — matrix operand order repaired and checked by reading, not compiled, not run on hardware
//
#include <hip/hip_runtime.h>
#include <hip/hip_bf16.h>
#include <math.h>

typedef __attribute__((ext_vector_type(16))) _Float16 v16h;
typedef __attribute__((ext_vector_type(8)))  _Float16 v8h;
typedef __attribute__((ext_vector_type(8)))  float    v8f;
typedef __attribute__((ext_vector_type(4)))  int      v4i;

#define NB 4
#define NS 2048
#define ND 512
#define NH 8
#define NDK 64
#define INV_SCALE 0.044194173824159216f   /* 1/sqrt(512) */
#define LN_EPS 1e-5f

// ---- optional gfx1250 async global->LDS path (guarded: falls back to sync copy) ----
#if defined(__has_builtin)
#if __has_builtin(__builtin_amdgcn_global_load_async_to_lds_b128) && \
    __has_builtin(__builtin_amdgcn_s_wait_asynccnt)
#define HAVE_ASYNC_LDS 1
#endif
#endif
#ifndef HAVE_ASYNC_LDS
#define HAVE_ASYNC_LDS 0
#endif

#define AS1 __attribute__((address_space(1)))
#define AS3 __attribute__((address_space(3)))

// copy 16 bytes global -> LDS (async when available)
__device__ __forceinline__ void stage16(const _Float16* g, _Float16* l) {
#if HAVE_ASYNC_LDS
  __builtin_amdgcn_global_load_async_to_lds_b128((AS1 v4i*)g, (AS3 v4i*)l, 0, 0);
#else
  *(v8h*)l = *(const v8h*)g;
#endif
}
__device__ __forceinline__ void stage_wait() {
#if HAVE_ASYNC_LDS
  __builtin_amdgcn_s_wait_asynccnt(0);
#endif
  __syncthreads();
}

// ---------- WMMA operand helpers (CDNA5 16x16x32 f16 layouts) ----------

__device__ __forceinline__ v16h pack16(v8h lo, v8h hh) {
  v16h r;
#pragma unroll
  for (int i = 0; i < 8; ++i) { r[i] = lo[i]; r[i + 8] = hh[i]; }
  return r;
}

__device__ __forceinline__ v8h cvt8(v8f x) {
  v8h r;
#pragma unroll
  for (int i = 0; i < 8; ++i) r[i] = (_Float16)x[i];
  return r;
}

// A 16x32: lane holds row M=lane%16; element e<8 -> K=k0+e+8*hi, e>=8 -> K=k0+e+8+8*hi
__device__ __forceinline__ v16h load_a_f16(const _Float16* __restrict__ row, int k0, int hi) {
  v8h lo = *(const v8h*)(row + k0 + 8 * hi);
  v8h hh = *(const v8h*)(row + k0 + 16 + 8 * hi);
  return pack16(lo, hh);
}
__device__ __forceinline__ v16h load_a_f32(const float* __restrict__ row, int k0, int hi) {
  v8f lo = *(const v8f*)(row + k0 + 8 * hi);
  v8f hh = *(const v8f*)(row + k0 + 16 + 8 * hi);
  return pack16(cvt8(lo), cvt8(hh));
}
// B 32x16: lane holds col N=lane%16; element e<8 -> K=k0+e+8*hi, e>=8 -> K=k0+e+8+8*hi
__device__ __forceinline__ v16h load_b_f16(const _Float16* __restrict__ row, int k0, int hi) {
  v8h lo = *(const v8h*)(row + k0 + 8 * hi);
  v8h hh = *(const v8h*)(row + k0 + 16 + 8 * hi);
  return pack16(lo, hh);
}
__device__ __forceinline__ v16h load_b_f32(const float* __restrict__ row, int k0, int hi) {
  v8f lo = *(const v8f*)(row + k0 + 8 * hi);
  v8f hh = *(const v8f*)(row + k0 + 16 + 8 * hi);
  return pack16(cvt8(lo), cvt8(hh));
}

__device__ __forceinline__ v8f wmma16(v16h a, v16h b, v8f c) {
  return __builtin_amdgcn_wmma_f32_16x16x32_f16(false, a, false, b, (short)0, c, false, false);
}

// ---------- Kernel 1: fused QKV projection (inline f32->f16 convert) ----------
// mode 0/1: out = head-major f16 [H,B,S,64]. mode 2: out = transposed f16 [H,B,64,S].
__global__ void proj_kernel(const float* __restrict__ X, const float* __restrict__ W,
                            const float* __restrict__ bias, _Float16* __restrict__ out,
                            int mode) {
  const int tile = blockIdx.x * 4 + (threadIdx.x >> 5);
  const int nt = tile & 31;          // 512/16 n-tiles
  const int mt = tile >> 5;          // 8192/16 m-tiles
  const int l = threadIdx.x & 31;
  const int col = l & 15;
  const int hi = l >> 4;

  const float* arow = X + (size_t)(mt * 16 + col) * ND;
  const int n = nt * 16 + col;
  const float* brow = W + (size_t)n * ND;   // torch Linear: y = x @ W.T -> B[k][n] = W[n][k]

  v8f acc = {};
#pragma unroll
  for (int c = 0; c < ND / 32; ++c) {
    v16h a = load_a_f32(arow, c * 32, hi);
    v16h b = load_b_f32(brow, c * 32, hi);
    acc = wmma16(a, b, acc);
  }
  const float bn = bias[n];
  const int h = n >> 6;
  const int d = n & 63;
#pragma unroll
  for (int r = 0; r < 8; ++r) {
    const int mg = mt * 16 + r + 8 * hi;
    const int bb = mg >> 11;   // / 2048
    const int s  = mg & 2047;
    const _Float16 v = (_Float16)(acc[r] + bn);
    if (mode == 2)
      out[(((size_t)h * NB + bb) * NDK + d) * NS + s] = v;   // [h,b,dv,s]
    else
      out[(((size_t)h * NB + bb) * NS + s) * NDK + d] = v;   // [h,b,s,d]
  }
}

// ---------- Kernel 2: scores + mask-quirk + softmax, att -> d_out ----------
// grid (S/16, H, B), block 256 = 8 waves; wave w owns key tiles w*256..w*256+255.
// Q tile (16x64 f16, shared by all 8 waves) staged through LDS (async on gfx1250).
__global__ void attn_kernel(const _Float16* __restrict__ qh, const _Float16* __restrict__ kh,
                            const unsigned char* __restrict__ mask, float* __restrict__ att) {
  const int qt = blockIdx.x;
  const int h  = blockIdx.y;
  const int b  = blockIdx.z;
  const int hb = h * NB + b;
  const int w  = threadIdx.x >> 5;
  const int l  = threadIdx.x & 31;
  const int col = l & 15;
  const int hi  = l >> 4;

  const _Float16* Q = qh + (size_t)hb * NS * NDK;
  const _Float16* K = kh + (size_t)hb * NS * NDK;

  // stage Q tile [16 x 64] into LDS (1024 halves = 128 x 16B chunks)
  __shared__ _Float16 sQ[16 * NDK];
  {
    const _Float16* qsrc = Q + (size_t)(qt * 16) * NDK;
    for (int i = threadIdx.x; i < (16 * NDK) / 8; i += 256)
      stage16(qsrc + i * 8, sQ + i * 8);
  }
  stage_wait();

  v8f acc[16];
  v8f z = {};
#pragma unroll
  for (int t = 0; t < 16; ++t) acc[t] = z;

#pragma unroll
  for (int c = 0; c < 2; ++c) {             // dk = 64 -> two K-chunks of 32
    v8h alo = *(const v8h*)(&sQ[col * NDK + c * 32 + 8 * hi]);
    v8h ahi = *(const v8h*)(&sQ[col * NDK + c * 32 + 16 + 8 * hi]);
    v16h a = pack16(alo, ahi);
#pragma unroll
    for (int t = 0; t < 16; ++t) {
      const int n = (w * 16 + t) * 16 + col;      // key index (B-matrix column)
      v16h bm = load_b_f16(K + (size_t)n * NDK, c * 32, hi);
      acc[t] = wmma16(a, bm, acc[t]);
    }
  }

  // scale, then faithful quirk: masked positions := value 1e-9 (not -inf)
  const unsigned char* mb = mask + (size_t)b * NS * NS;
#pragma unroll
  for (int t = 0; t < 16; ++t) {
    const int n = (w * 16 + t) * 16 + col;
#pragma unroll
    for (int r = 0; r < 8; ++r) {
      const int m = qt * 16 + r + 8 * hi;
      float v = acc[t][r] * INV_SCALE;
      if (mb[(size_t)m * NS + n]) v = 1e-9f;
      acc[t][r] = v;
    }
  }

  // softmax over 2048 keys/row: shuffle within 16-lane groups, LDS across 8 waves
  __shared__ float sred[8][16];
  float rmax[8];
#pragma unroll
  for (int r = 0; r < 8; ++r) {
    float m = -3.0e38f;
#pragma unroll
    for (int t = 0; t < 16; ++t) m = fmaxf(m, acc[t][r]);
#pragma unroll
    for (int d = 1; d < 16; d <<= 1) m = fmaxf(m, __shfl_xor(m, d, 32));
    rmax[r] = m;
  }
  if (col == 0) {
#pragma unroll
    for (int r = 0; r < 8; ++r) sred[w][r + 8 * hi] = rmax[r];
  }
  __syncthreads();
#pragma unroll
  for (int r = 0; r < 8; ++r) {
    const int row = r + 8 * hi;
    float m = sred[0][row];
#pragma unroll
    for (int ww = 1; ww < 8; ++ww) m = fmaxf(m, sred[ww][row]);
    rmax[r] = m;
  }
  __syncthreads();

  float rsum[8];
#pragma unroll
  for (int r = 0; r < 8; ++r) rsum[r] = 0.f;
#pragma unroll
  for (int t = 0; t < 16; ++t) {
#pragma unroll
    for (int r = 0; r < 8; ++r) {
      const float p = __expf(acc[t][r] - rmax[r]);
      acc[t][r] = p;
      rsum[r] += p;
    }
  }
#pragma unroll
  for (int r = 0; r < 8; ++r) {
#pragma unroll
    for (int d = 1; d < 16; d <<= 1) rsum[r] += __shfl_xor(rsum[r], d, 32);
  }
  if (col == 0) {
#pragma unroll
    for (int r = 0; r < 8; ++r) sred[w][r + 8 * hi] = rsum[r];
  }
  __syncthreads();
#pragma unroll
  for (int r = 0; r < 8; ++r) {
    const int row = r + 8 * hi;
    float s = 0.f;
#pragma unroll
    for (int ww = 0; ww < 8; ++ww) s += sred[ww][row];
    rsum[r] = 1.0f / s;
  }

  float* arow = att + (size_t)hb * NS * NS;
#pragma unroll
  for (int t = 0; t < 16; ++t) {
    const int n = (w * 16 + t) * 16 + col;
#pragma unroll
    for (int r = 0; r < 8; ++r) {
      const int m = qt * 16 + r + 8 * hi;
      arow[(size_t)m * NS + n] = acc[t][r] * rsum[r];
    }
  }
}

// ---------- Kernel 3: out_h = att @ V ----------
// grid (S/64, H*B), block 128 = 4 waves; wave owns one m-tile and ALL 4 dv-tiles,
// so each element of att (the HBM-resident 537MB tensor) is read exactly once.
// __builtin_prefetch -> global_prefetch_b8 hides HBM latency on the att stream.
__global__ void attv_kernel(const float* __restrict__ att, const _Float16* __restrict__ vhT,
                            _Float16* __restrict__ attout) {
  const int w  = threadIdx.x >> 5;
  const int mt = blockIdx.x * 4 + w;
  const int hb = blockIdx.y;
  const int h = hb >> 2;   // B = 4
  const int b = hb & 3;
  const int l = threadIdx.x & 31;
  const int col = l & 15;
  const int hi  = l >> 4;

  const float*    arow  = att + (size_t)hb * NS * NS + (size_t)(mt * 16 + col) * NS;
  const _Float16* vbase = vhT + (size_t)hb * NDK * NS;

  v8f acc[4];
  v8f z = {};
#pragma unroll
  for (int j = 0; j < 4; ++j) acc[j] = z;

  for (int c = 0; c < NS / 32; ++c) {       // K = 2048 -> 64 chunks
    const int k0 = c * 32;
    if (c < (NS / 32) - 8)
      __builtin_prefetch(arow + k0 + 256, 0, 1);   // 8 chunks ahead on the att stream
    v16h a = load_a_f32(arow, k0, hi);
#pragma unroll
    for (int j = 0; j < 4; ++j) {
      const _Float16* vrow = vbase + (size_t)(j * 16 + col) * NS;
      v16h bm = load_b_f16(vrow, k0, hi);
      acc[j] = wmma16(a, bm, acc[j]);
    }
  }
#pragma unroll
  for (int j = 0; j < 4; ++j) {
#pragma unroll
    for (int r = 0; r < 8; ++r) {
      const int s = mt * 16 + r + 8 * hi;
      attout[((size_t)(b * NS + s)) * ND + h * NDK + j * 16 + col] = (_Float16)acc[j][r];
    }
  }
}

// ---------- Kernel 4: y = LN((att_out @ Wo.T + bo) + residual) ----------
// grid (8192/16), block 256 = 8 waves; wave owns 4 n-tiles (full 512-wide row in WG).
// Shared 16x512 A tile staged once through LDS (async on gfx1250) instead of 8x from L2.
__global__ void final_kernel(const _Float16* __restrict__ attout, const float* __restrict__ Wo,
                             const float* __restrict__ bo, const float* __restrict__ resid,
                             const float* __restrict__ g, const float* __restrict__ beta,
                             float* __restrict__ y) {
  const int mt = blockIdx.x;
  const int w  = threadIdx.x >> 5;
  const int l  = threadIdx.x & 31;
  const int col = l & 15;
  const int hi  = l >> 4;

  __shared__ _Float16 sA[16 * ND];   // 16 KB
  {
    const _Float16* src = attout + (size_t)(mt * 16) * ND;
    for (int i = threadIdx.x; i < (16 * ND) / 8; i += 256)
      stage16(src + i * 8, sA + i * 8);
  }
  stage_wait();

  v8f acc[4];
  v8f z = {};
#pragma unroll
  for (int j = 0; j < 4; ++j) acc[j] = z;

#pragma unroll
  for (int c = 0; c < ND / 32; ++c) {
    v8h alo = *(const v8h*)(&sA[col * ND + c * 32 + 8 * hi]);
    v8h ahi = *(const v8h*)(&sA[col * ND + c * 32 + 16 + 8 * hi]);
    v16h a = pack16(alo, ahi);
#pragma unroll
    for (int j = 0; j < 4; ++j) {
      const int n = (w * 4 + j) * 16 + col;
      v16h bm = load_b_f32(Wo + (size_t)n * ND, c * 32, hi);
      acc[j] = wmma16(a, bm, acc[j]);
    }
  }

  float yv[4][8];
#pragma unroll
  for (int j = 0; j < 4; ++j) {
    const int n = (w * 4 + j) * 16 + col;
    const float bn = bo[n];
#pragma unroll
    for (int r = 0; r < 8; ++r) {
      const int m = mt * 16 + r + 8 * hi;
      yv[j][r] = acc[j][r] + bn + resid[(size_t)m * ND + n];
    }
  }

  float ps[8], pq[8];
#pragma unroll
  for (int r = 0; r < 8; ++r) { ps[r] = 0.f; pq[r] = 0.f; }
#pragma unroll
  for (int j = 0; j < 4; ++j) {
#pragma unroll
    for (int r = 0; r < 8; ++r) { ps[r] += yv[j][r]; pq[r] += yv[j][r] * yv[j][r]; }
  }
#pragma unroll
  for (int r = 0; r < 8; ++r) {
#pragma unroll
    for (int d = 1; d < 16; d <<= 1) {
      ps[r] += __shfl_xor(ps[r], d, 32);
      pq[r] += __shfl_xor(pq[r], d, 32);
    }
  }
  __shared__ float s1[8][16], s2[8][16];
  if (col == 0) {
#pragma unroll
    for (int r = 0; r < 8; ++r) { s1[w][r + 8 * hi] = ps[r]; s2[w][r + 8 * hi] = pq[r]; }
  }
  __syncthreads();
  float mean[8], rstd[8];
#pragma unroll
  for (int r = 0; r < 8; ++r) {
    const int row = r + 8 * hi;
    float sum = 0.f, sq = 0.f;
#pragma unroll
    for (int ww = 0; ww < 8; ++ww) { sum += s1[ww][row]; sq += s2[ww][row]; }
    const float mu = sum * (1.0f / ND);
    mean[r] = mu;
    rstd[r] = rsqrtf(sq * (1.0f / ND) - mu * mu + LN_EPS);
  }
#pragma unroll
  for (int j = 0; j < 4; ++j) {
    const int n = (w * 4 + j) * 16 + col;
    const float gn = g[n], bn = beta[n];
#pragma unroll
    for (int r = 0; r < 8; ++r) {
      const int m = mt * 16 + r + 8 * hi;
      y[(size_t)m * ND + n] = (yv[j][r] - mean[r]) * rstd[r] * gn + bn;
    }
  }
}

// ---------- launch ----------
extern "C" void kernel_launch(void* const* d_in, const int* in_sizes, int n_in,
                              void* d_out, int out_size, void* d_ws, size_t ws_size,
                              hipStream_t stream) {
  (void)in_sizes; (void)n_in; (void)out_size; (void)ws_size;
  const float* q  = (const float*)d_in[0];
  const float* k  = (const float*)d_in[1];
  const float* v  = (const float*)d_in[2];
  const unsigned char* mask = (const unsigned char*)d_in[3];  // jax bool = 1 byte
  const float* Wq = (const float*)d_in[4];
  const float* bq = (const float*)d_in[5];
  const float* Wk = (const float*)d_in[6];
  const float* bk = (const float*)d_in[7];
  const float* Wv = (const float*)d_in[8];
  const float* bv = (const float*)d_in[9];
  const float* Wo = (const float*)d_in[10];
  const float* bo = (const float*)d_in[11];
  const float* ln_g = (const float*)d_in[12];
  const float* ln_b = (const float*)d_in[13];

  float* y   = (float*)d_out;                       // [B,S,D]
  float* att = y + (size_t)NB * NS * ND;            // [H,B,S,S]

  const size_t headElems = (size_t)NH * NB * NS * NDK;   // 4,194,304 halves each
  _Float16* qh     = (_Float16*)d_ws;               // [H,B,S,64]
  _Float16* kh     = qh + headElems;                // [H,B,S,64]
  _Float16* vhT    = kh + headElems;                // [H,B,64,S]
  _Float16* attout = vhT + headElems;               // [B,S,H*64]

  const int projBlocks = (NB * NS / 16) * (ND / 16) / 4;  // 4096
  proj_kernel<<<dim3(projBlocks), dim3(128), 0, stream>>>(q, Wq, bq, qh, 0);
  proj_kernel<<<dim3(projBlocks), dim3(128), 0, stream>>>(k, Wk, bk, kh, 1);
  proj_kernel<<<dim3(projBlocks), dim3(128), 0, stream>>>(v, Wv, bv, vhT, 2);

  attn_kernel<<<dim3(NS / 16, NH, NB), dim3(256), 0, stream>>>(qh, kh, mask, att);
  attv_kernel<<<dim3(NS / 64, NH * NB), dim3(128), 0, stream>>>(att, vhT, attout);
  final_kernel<<<dim3(NB * NS / 16), dim3(256), 0, stream>>>(attout, Wo, bo, q, ln_g, ln_b, y);
}
